// FoldNet_Encoder_11665131176431
// MI455X (gfx1250) — matrix operand order repaired and checked by reading, not compiled, not run on hardware
//
#include <hip/hip_runtime.h>
#include <hip/hip_bf16.h>

// ---------------------------------------------------------------------------
// FoldNet encoder for MI455X (gfx1250, wave32, WMMA).
// Channel-mixing GEMMs use v_wmma_f32_16x16x32_f16 (f16 in, f32 acc).
// Each wave computes a 16x64 output strip (4 N-tiles): one A fragment is
// reused across 4 B fragments -> 4 chained WMMAs per K-step (4x less A
// traffic than one-tile-per-wave). 4 waves per block cover a 64x64 macro-tile.
// ---------------------------------------------------------------------------

typedef __attribute__((ext_vector_type(16))) _Float16 v16h;
typedef __attribute__((ext_vector_type(8)))  _Float16 v8h;
typedef __attribute__((ext_vector_type(8)))  float    v8f;

#define BATCH 8
#define NPTS  2048
#define KNEI  16
#define MTOT  (BATCH * NPTS)   // 16384 points

// ---------------------------------------------------------------------------
// 1) kNN: per query point, top-16 of neg_dist = 2<q,p> - |q|^2 - |p|^2.
//    Candidate points tiled through LDS; fully-unrolled bubble insertion
//    keeps the sorted list in registers. Strict '>' keeps earliest index on
//    ties (matches lax.top_k ordering; self with nd==0 lands first).
// ---------------------------------------------------------------------------
__global__ void knn_kernel(const float* __restrict__ pts, int* __restrict__ idx) {
    __shared__ float tile[256 * 3];
    const int b   = blockIdx.y;
    const int n   = blockIdx.x * 256 + threadIdx.x;
    const float* pb = pts + (size_t)b * NPTS * 3;

    const float qx = pb[n * 3 + 0], qy = pb[n * 3 + 1], qz = pb[n * 3 + 2];
    const float sqq = qx * qx + qy * qy + qz * qz;

    float bd[KNEI];
    int   bi[KNEI];
#pragma unroll
    for (int t = 0; t < KNEI; ++t) { bd[t] = -3.4e38f; bi[t] = 0; }

    for (int t0 = 0; t0 < NPTS; t0 += 256) {
        __syncthreads();
        const int j = t0 + threadIdx.x;
        tile[threadIdx.x * 3 + 0] = pb[j * 3 + 0];
        tile[threadIdx.x * 3 + 1] = pb[j * 3 + 1];
        tile[threadIdx.x * 3 + 2] = pb[j * 3 + 2];
        __syncthreads();
        for (int jj = 0; jj < 256; ++jj) {
            const float px = tile[jj * 3 + 0];
            const float py = tile[jj * 3 + 1];
            const float pz = tile[jj * 3 + 2];
            const float sqp = px * px + py * py + pz * pz;
            const float nd  = 2.0f * (qx * px + qy * py + qz * pz) - sqq - sqp;
            if (nd > bd[KNEI - 1]) {
                bd[KNEI - 1] = nd; bi[KNEI - 1] = t0 + jj;
#pragma unroll
                for (int t = KNEI - 1; t > 0; --t) {
                    if (bd[t] > bd[t - 1]) {
                        float fd = bd[t]; bd[t] = bd[t - 1]; bd[t - 1] = fd;
                        int   fi = bi[t]; bi[t] = bi[t - 1]; bi[t - 1] = fi;
                    }
                }
            }
        }
    }
    int* op = idx + ((size_t)b * NPTS + n) * KNEI;
#pragma unroll
    for (int t = 0; t < KNEI; ++t) op[t] = bi[t];
}

// ---------------------------------------------------------------------------
// 2) x0 = concat(pts(3), cov(9)) padded to 32 channels, stored f16 (M,32).
//    cov = outer(neigh0, neigh1), row-major 3x3.
// ---------------------------------------------------------------------------
__global__ void build_x0_kernel(const float* __restrict__ pts,
                                const int* __restrict__ idx,
                                _Float16* __restrict__ x0) {
    const int t = blockIdx.x * 256 + threadIdx.x;
    if (t >= MTOT) return;
    const int b = t / NPTS;
    const float* pb = pts + (size_t)b * NPTS * 3;
    const int n = t % NPTS;
    const int i0 = idx[(size_t)t * KNEI + 0];
    const int i1 = idx[(size_t)t * KNEI + 1];
    const float* p  = pb + n  * 3;
    const float* q0 = pb + i0 * 3;
    const float* q1 = pb + i1 * 3;
    _Float16* o = x0 + (size_t)t * 32;
    o[0] = (_Float16)p[0]; o[1] = (_Float16)p[1]; o[2] = (_Float16)p[2];
#pragma unroll
    for (int i = 0; i < 3; ++i)
#pragma unroll
        for (int j = 0; j < 3; ++j)
            o[3 + 3 * i + j] = (_Float16)(q0[i] * q1[j]);
#pragma unroll
    for (int c = 12; c < 32; ++c) o[c] = (_Float16)0.0f;
}

// ---------------------------------------------------------------------------
// 3) f32 -> f16 weight conversion, zero-padded along K.  src (O,Cin) row-major.
// ---------------------------------------------------------------------------
__global__ void cvt_weight_kernel(const float* __restrict__ src,
                                  _Float16* __restrict__ dst,
                                  int O, int Cin, int Kpad) {
    const int t = blockIdx.x * 256 + threadIdx.x;
    if (t >= O * Kpad) return;
    const int o = t / Kpad, k = t % Kpad;
    dst[t] = (k < Cin) ? (_Float16)src[(size_t)o * Cin + k] : (_Float16)0.0f;
}

// ---------------------------------------------------------------------------
// 4) WMMA GEMM + bias (+ReLU):  Y(M,O) = act( X(M,Kd) * W(O,Kd)^T + bias ).
//    Wave strip = 16 rows x 64 cols (4 accumulators); block = 4 waves
//    covering a 64x64 macro-tile.  Per K-step: 1 A fragment, 4 B fragments,
//    4 chained v_wmma_f32_16x16x32_f16.
//    Fragment layouts used by this kernel (wave32):
//      A: lane l -> row M = l&15;  K halves = [16g,16g+16)
//      B: lane l -> col N = l&15;  K halves = [16g,16g+16)   (g = l>>4)
//      D: vgpr r -> M = r + 8g, N = l&15
// ---------------------------------------------------------------------------
template <int ACT>
__global__ void wmma_gemm_kernel(const _Float16* __restrict__ X,
                                 const _Float16* __restrict__ W,
                                 const float* __restrict__ bias,
                                 _Float16* __restrict__ Y,
                                 int Kd, int O) {
    const int lane  = threadIdx.x & 31;
    const int wave  = threadIdx.x >> 5;          // 0..3
    const int g     = lane >> 4;
    const int lr    = lane & 15;
    const int mBase = blockIdx.x * 64 + wave * 16;
    const int oBase = blockIdx.y * 64;

    const _Float16* xrow = X + (size_t)(mBase + lr) * Kd;   // A row (M = lr)
    const _Float16* wrow = W + (size_t)(oBase + lr) * Kd;   // B col tile 0
    const size_t wstride = (size_t)16 * Kd;                 // next 16-col tile

    v8f acc0 = {}, acc1 = {}, acc2 = {}, acc3 = {};
    for (int k0 = 0; k0 < Kd; k0 += 32) {
        __builtin_prefetch((const void*)(xrow + k0 + 32), 0, 0);  // global_prefetch_b8
        __builtin_prefetch((const void*)(wrow + k0 + 32), 0, 0);

        const v16h a = *(const v16h*)(xrow + k0 + 16 * g);       // K = 16g..16g+15

        const v16h b0 = *(const v16h*)(wrow + 0 * wstride + k0 + 16 * g);
        const v16h b1 = *(const v16h*)(wrow + 1 * wstride + k0 + 16 * g);
        const v16h b2 = *(const v16h*)(wrow + 2 * wstride + k0 + 16 * g);
        const v16h b3 = *(const v16h*)(wrow + 3 * wstride + k0 + 16 * g);

        acc0 = __builtin_amdgcn_wmma_f32_16x16x32_f16(false, a, false, b0,
                                                      (short)0, acc0, false, false);
        acc1 = __builtin_amdgcn_wmma_f32_16x16x32_f16(false, a, false, b1,
                                                      (short)0, acc1, false, false);
        acc2 = __builtin_amdgcn_wmma_f32_16x16x32_f16(false, a, false, b2,
                                                      (short)0, acc2, false, false);
        acc3 = __builtin_amdgcn_wmma_f32_16x16x32_f16(false, a, false, b3,
                                                      (short)0, acc3, false, false);
    }

    // Epilogue: bias + optional ReLU, store f16 strip (D: M = r + 8g, N = lr)
    v8f accs[4] = {acc0, acc1, acc2, acc3};
#pragma unroll
    for (int j = 0; j < 4; ++j) {
        const int col = oBase + 16 * j + lr;
        const float bcol = bias[col];
#pragma unroll
        for (int r = 0; r < 8; ++r) {
            float v = accs[j][r] + bcol;
            if (ACT) v = fmaxf(v, 0.0f);
            Y[(size_t)(mBase + r + 8 * g) * O + col] = (_Float16)v;
        }
    }
}

// ---------------------------------------------------------------------------
// 5) Local max-pool over 16 gathered neighbors, vectorized 8 channels/thread:
//    P[m, c0:c0+8] = max_k A[b, idx[m,k], c0:c0+8]   (one b128 load per k)
// ---------------------------------------------------------------------------
__global__ void local_maxpool_kernel(const _Float16* __restrict__ A,
                                     const int* __restrict__ idx,
                                     _Float16* __restrict__ P, int C) {
    const int chunks = C >> 3;                       // 8 channels per thread
    const int t = blockIdx.x * 256 + threadIdx.x;
    if (t >= MTOT * chunks) return;
    const int ch = (t % chunks) << 3;
    const int m  = t / chunks;
    const int b  = m / NPTS;
    const int* ip = idx + (size_t)m * KNEI;
    float best[8];
#pragma unroll
    for (int i = 0; i < 8; ++i) best[i] = -3.4e38f;
#pragma unroll
    for (int k = 0; k < KNEI; ++k) {
        const v8h v = *(const v8h*)(A + ((size_t)b * NPTS + ip[k]) * C + ch);
#pragma unroll
        for (int i = 0; i < 8; ++i) best[i] = fmaxf(best[i], (float)v[i]);
    }
    v8h o;
#pragma unroll
    for (int i = 0; i < 8; ++i) o[i] = (_Float16)best[i];
    *(v8h*)(P + (size_t)m * C + ch) = o;
}

// ---------------------------------------------------------------------------
// 6) Global max-pool over N: G[b,c] = max_n A[b,n,c]  (C = 1024, f32 out)
// ---------------------------------------------------------------------------
__global__ void global_maxpool_kernel(const _Float16* __restrict__ A,
                                      float* __restrict__ G) {
    const int t = blockIdx.x * 256 + threadIdx.x;
    if (t >= BATCH * 1024) return;
    const int c = t % 1024;
    const int b = t / 1024;
    float best = -3.4e38f;
    for (int n = 0; n < NPTS; ++n)
        best = fmaxf(best, (float)A[((size_t)b * NPTS + n) * 1024 + c]);
    G[t] = best;
}

// ---------------------------------------------------------------------------
// 7) mlp2 head (M=8, f32 scalar): H = relu(G*w2a^T+b2a); out = H*w2b^T+b2b
// ---------------------------------------------------------------------------
__global__ void mlp2a_kernel(const float* __restrict__ G,
                             const float* __restrict__ w, const float* __restrict__ bias,
                             float* __restrict__ H) {
    const int t = blockIdx.x * 256 + threadIdx.x;
    if (t >= BATCH * 512) return;
    const int o = t % 512, b = t / 512;
    float s = bias[o];
    const float* gp = G + (size_t)b * 1024;
    const float* wp = w + (size_t)o * 1024;
    for (int c = 0; c < 1024; ++c) s += gp[c] * wp[c];
    H[t] = fmaxf(s, 0.0f);
}

__global__ void mlp2b_kernel(const float* __restrict__ H,
                             const float* __restrict__ w, const float* __restrict__ bias,
                             float* __restrict__ out) {
    const int t = blockIdx.x * 256 + threadIdx.x;
    if (t >= BATCH * 512) return;
    const int o = t % 512, b = t / 512;
    float s = bias[o];
    const float* hp = H + (size_t)b * 512;
    const float* wp = w + (size_t)o * 512;
    for (int c = 0; c < 512; ++c) s += hp[c] * wp[c];
    out[t] = s;
}

// ---------------------------------------------------------------------------
// Launch
// ---------------------------------------------------------------------------
extern "C" void kernel_launch(void* const* d_in, const int* in_sizes, int n_in,
                              void* d_out, int out_size, void* d_ws, size_t ws_size,
                              hipStream_t stream) {
    (void)in_sizes; (void)n_in; (void)out_size; (void)ws_size;

    const float* pts     = (const float*)d_in[0];
    const float* w1a     = (const float*)d_in[1];  const float* b1a = (const float*)d_in[2];
    const float* w1b     = (const float*)d_in[3];  const float* b1b = (const float*)d_in[4];
    const float* w1c     = (const float*)d_in[5];  const float* b1c = (const float*)d_in[6];
    const float* lin1_w  = (const float*)d_in[7];  const float* lin1_b = (const float*)d_in[8];
    const float* conv1_w = (const float*)d_in[9];  const float* conv1_b = (const float*)d_in[10];
    const float* lin2_w  = (const float*)d_in[11]; const float* lin2_b = (const float*)d_in[12];
    const float* conv2_w = (const float*)d_in[13]; const float* conv2_b = (const float*)d_in[14];
    const float* w2a     = (const float*)d_in[15]; const float* b2a = (const float*)d_in[16];
    const float* w2b     = (const float*)d_in[17]; const float* b2b = (const float*)d_in[18];
    float* out = (float*)d_out;

    // Workspace carve-up (256B aligned regions)
    size_t off = 0;
    char* base = (char*)d_ws;
    auto carve = [&](size_t bytes) -> char* {
        char* p = base + off;
        off += (bytes + 255) & ~(size_t)255;
        return p;
    };
    int*      idx   = (int*)     carve((size_t)MTOT * KNEI * 4);   //  1 MB
    _Float16* x0    = (_Float16*)carve((size_t)MTOT * 32 * 2);     //  1 MB
    _Float16* buf0  = (_Float16*)carve((size_t)MTOT * 128 * 2);    //  4 MB (ping)
    _Float16* buf1  = (_Float16*)carve((size_t)MTOT * 128 * 2);    //  4 MB (pong)
    _Float16* a7    = (_Float16*)carve((size_t)MTOT * 1024 * 2);   // 32 MB
    float*    gpool = (float*)   carve((size_t)BATCH * 1024 * 4);
    float*    hmid  = (float*)   carve((size_t)BATCH * 512 * 4);
    _Float16* w1a_h   = (_Float16*)carve(64 * 32 * 2);
    _Float16* w1b_h   = (_Float16*)carve(64 * 64 * 2);
    _Float16* w1c_h   = (_Float16*)carve(64 * 64 * 2);
    _Float16* lin1_h  = (_Float16*)carve(64 * 64 * 2);
    _Float16* conv1_h = (_Float16*)carve(128 * 64 * 2);
    _Float16* lin2_h  = (_Float16*)carve(128 * 128 * 2);
    _Float16* conv2_h = (_Float16*)carve(1024 * 128 * 2);

    // --- weight conversions (f32 -> f16, pad K of first layer 12 -> 32) ---
    auto cvt = [&](const float* s, _Float16* d, int O, int Cin, int Kpad) {
        const int tot = O * Kpad;
        cvt_weight_kernel<<<(tot + 255) / 256, 256, 0, stream>>>(s, d, O, Cin, Kpad);
    };
    cvt(w1a,     w1a_h,   64,   12,  32);
    cvt(w1b,     w1b_h,   64,   64,  64);
    cvt(w1c,     w1c_h,   64,   64,  64);
    cvt(lin1_w,  lin1_h,  64,   64,  64);
    cvt(conv1_w, conv1_h, 128,  64,  64);
    cvt(lin2_w,  lin2_h,  128, 128, 128);
    cvt(conv2_w, conv2_h, 1024,128, 128);

    // --- kNN + feature build ---
    knn_kernel<<<dim3(NPTS / 256, BATCH), 256, 0, stream>>>(pts, idx);
    build_x0_kernel<<<(MTOT + 255) / 256, 256, 0, stream>>>(pts, idx, x0);

    const dim3 wblk(128, 1, 1);   // 4 waves -> 64x64 macro-tile
    auto gemm = [&](const _Float16* X, const _Float16* W, const float* bias,
                    _Float16* Y, int Kd, int O, bool relu) {
        const dim3 grid(MTOT / 64, O / 64);
        if (relu) wmma_gemm_kernel<1><<<grid, wblk, 0, stream>>>(X, W, bias, Y, Kd, O);
        else      wmma_gemm_kernel<0><<<grid, wblk, 0, stream>>>(X, W, bias, Y, Kd, O);
    };

    // mlp1: 12(->32) -> 64 -> 64 -> 64, all ReLU
    gemm(x0,   w1a_h, b1a, buf0, 32, 64, true);
    gemm(buf0, w1b_h, b1b, buf1, 64, 64, true);
    gemm(buf1, w1c_h, b1c, buf0, 64, 64, true);

    // graph layer 1: local maxpool -> lin1 (no act) -> conv1 (ReLU)
    local_maxpool_kernel<<<(MTOT * 64 / 8 + 255) / 256, 256, 0, stream>>>(buf0, idx, buf1, 64);
    gemm(buf1, lin1_h,  lin1_b,  buf0, 64, 64,  false);
    gemm(buf0, conv1_h, conv1_b, buf1, 64, 128, true);

    // graph layer 2: local maxpool -> lin2 (no act) -> conv2 (no act, 1024ch)
    local_maxpool_kernel<<<(MTOT * 128 / 8 + 255) / 256, 256, 0, stream>>>(buf1, idx, buf0, 128);
    gemm(buf0, lin2_h,  lin2_b,  buf1, 128, 128,  false);
    gemm(buf1, conv2_h, conv2_b, a7,   128, 1024, false);

    // global max pool + mlp2 head
    global_maxpool_kernel<<<(BATCH * 1024 + 255) / 256, 256, 0, stream>>>(a7, gpool);
    mlp2a_kernel<<<(BATCH * 512 + 255) / 256, 256, 0, stream>>>(gpool, w2a, b2a, hmid);
    mlp2b_kernel<<<(BATCH * 512 + 255) / 256, 256, 0, stream>>>(hmid, w2b, b2b, out);
}
